// DeformableTemporalEncoder_89618787598352
// MI455X (gfx1250) — matrix operand order repaired and checked by reading, not compiled, not run on hardware
//
#include <hip/hip_runtime.h>
#include <hip/hip_bf16.h>
#include <cstdint>

typedef __attribute__((ext_vector_type(16))) _Float16 v16h;
typedef __attribute__((ext_vector_type(8)))  _Float16 v8h;
typedef __attribute__((ext_vector_type(8)))  float    v8f;

#define B_CLIPS  1024
#define T_LEN    14
#define DM       1024
#define NHEADS   16
#define NPOINTS  4
#define HD       64
#define M_ROWS   (B_CLIPS * T_LEN)   // 14336

#define BM  128
#define BN  64
#define BK  32
#define LDA 48   // padded half-stride: 96 bytes -> every 16B chunk stays 16B-aligned
#define LDB 48

// low 32 bits of a flat shared-pointer == LDS byte offset
__device__ __forceinline__ unsigned lds_off(const void* p) {
    return (unsigned)(uintptr_t)p;
}

// ---------------------------------------------------------------------------
// fp32 -> fp16 conversion (straight copy, for x)
// ---------------------------------------------------------------------------
__global__ void f32_to_f16_kernel(const float* __restrict__ src,
                                  _Float16* __restrict__ dst, int n) {
    int i = blockIdx.x * blockDim.x + threadIdx.x;
    if (i < n) dst[i] = (_Float16)src[i];
}

// ---------------------------------------------------------------------------
// fp32 [K,N] -> fp16 transposed [N,K]   (weights; done once per launch)
// ---------------------------------------------------------------------------
__global__ void transpose_cvt_kernel(const float* __restrict__ src,
                                     _Float16* __restrict__ dst,
                                     int K, int N) {
    __shared__ float tile[32][33];
    int n0 = blockIdx.x * 32;
    int k0 = blockIdx.y * 32;
    #pragma unroll
    for (int i = threadIdx.y; i < 32; i += 8)
        tile[i][threadIdx.x] = src[(size_t)(k0 + i) * N + n0 + threadIdx.x];
    __syncthreads();
    #pragma unroll
    for (int i = threadIdx.y; i < 32; i += 8)
        dst[(size_t)(n0 + i) * K + k0 + threadIdx.x] = (_Float16)tile[threadIdx.x][i];
}

// ---------------------------------------------------------------------------
// WMMA GEMM: C[M,N] = A[M,K](f16) @ BT[N,K](f16)^T + bias[N], fp32 accumulate.
// Block tile 128x64, 8 waves, each wave a 32x32 region = 2x2 wmma accumulators.
// Tiles staged with GLOBAL_LOAD_ASYNC_TO_LDS_B128 (ASYNCcnt), double-buffered.
// Pipeline per K step: issue async stage(k+1) -> ds_load frags(k) -> wmma(k)
//                      -> sched_barrier -> s_wait_asynccnt -> s_barrier.
// M%128==0, N%64==0, K%32==0 (exact for this problem).
// ---------------------------------------------------------------------------
template <bool STORE_F16>
__global__ __launch_bounds__(256, 2)
void wmma_gemm_kernel(const _Float16* __restrict__ A,
                      const _Float16* __restrict__ BT,
                      const float* __restrict__ bias,
                      void* __restrict__ Cout,
                      int M, int N, int K)
{
    __shared__ _Float16 As[2][BM * LDA];
    __shared__ _Float16 Bs[2][BN * LDB];   // [n][k]

    const int tid   = threadIdx.x;
    const int lane  = tid & 31;
    const int wid   = tid >> 5;
    const int waveM = wid & 3;          // 0..3
    const int waveN = wid >> 2;         // 0..1
    const int m0 = blockIdx.y * BM;
    const int n0 = blockIdx.x * BN;

    const int lrow  = lane & 15;        // M/N position within 16
    const int khalf = lane >> 4;        // K-half selector

    v8f acc[2][2] = {};

    // async-stage one K tile into LDS buffer `buf`
    auto stage = [&](int buf, int kt) {
        // A tile: 128 rows x 32 halves = 512 x 16B chunks; 2 per thread
        #pragma unroll
        for (int c = 0; c < 2; ++c) {
            int chunk = tid + c * 256;
            int row = chunk >> 2;
            int kc  = (chunk & 3) << 3;
            const _Float16* g = A + (size_t)(m0 + row) * K + kt + kc;
            unsigned l = lds_off(&As[buf][row * LDA + kc]);
            asm volatile("global_load_async_to_lds_b128 %0, %1, off"
                         :: "v"(l), "v"(g) : "memory");
        }
        // B tile: 64 rows x 32 halves = 256 x 16B chunks; 1 per thread
        {
            int nrow = tid >> 2;
            int kc   = (tid & 3) << 3;
            const _Float16* g = BT + (size_t)(n0 + nrow) * K + kt + kc;
            unsigned l = lds_off(&Bs[buf][nrow * LDB + kc]);
            asm volatile("global_load_async_to_lds_b128 %0, %1, off"
                         :: "v"(l), "v"(g) : "memory");
        }
    };

    stage(0, 0);
    asm volatile("s_wait_asynccnt 0x0" ::: "memory");
    __syncthreads();

    for (int kt = 0; kt < K; kt += BK) {
        const int buf = (kt / BK) & 1;
        if (kt + BK < K) stage(buf ^ 1, kt + BK);   // overlap with WMMA below

        // ---- fragments ----
        // A lane layout (16-bit A 16x32): lane holds row = lane%16,
        // Ks = {klo..klo+7, klo+16..klo+23}, klo = (lane>>4)*8
        v16h af[2], bf[2];
        #pragma unroll
        for (int mi = 0; mi < 2; ++mi) {
            int r = waveM * 32 + mi * 16 + lrow;
            v8h lo = *(const v8h*)(&As[buf][r * LDA + khalf * 8]);
            v8h hi = *(const v8h*)(&As[buf][r * LDA + khalf * 8 + 16]);
            #pragma unroll
            for (int j = 0; j < 8; ++j) { af[mi][j] = lo[j]; af[mi][8 + j] = hi[j]; }
        }
        // B lane layout (32x16): lane holds col = lane%16, Ks = {klo..klo+7, klo+16..klo+23}, klo = (lane>>4)*8
        #pragma unroll
        for (int ni = 0; ni < 2; ++ni) {
            int ccol = waveN * 32 + ni * 16 + lrow;
            v8h lo = *(const v8h*)(&Bs[buf][ccol * LDB + khalf * 8]);
            v8h hi = *(const v8h*)(&Bs[buf][ccol * LDB + khalf * 8 + 16]);
            #pragma unroll
            for (int j = 0; j < 8; ++j) { bf[ni][j] = lo[j]; bf[ni][8 + j] = hi[j]; }
        }

        #pragma unroll
        for (int mi = 0; mi < 2; ++mi)
            #pragma unroll
            for (int ni = 0; ni < 2; ++ni)
                acc[mi][ni] = __builtin_amdgcn_wmma_f32_16x16x32_f16(
                    false, af[mi], false, bf[ni], (short)0, acc[mi][ni],
                    false, false);

        // keep the asynccnt wait AFTER the WMMA block so next-tile global->LDS
        // latency hides behind the matrix math, not just the ds_loads
        __builtin_amdgcn_sched_barrier(0);

        // own async loads landed in LDS, then workgroup-wide visibility
        asm volatile("s_wait_asynccnt 0x0" ::: "memory");
        __syncthreads();
    }

    // ---- epilogue: C lane layout: col = lane%16, VGPR v -> row v + 8*(lane>>4)
    #pragma unroll
    for (int mi = 0; mi < 2; ++mi) {
        #pragma unroll
        for (int ni = 0; ni < 2; ++ni) {
            int col = n0 + waveN * 32 + ni * 16 + lrow;
            float bv = bias[col];
            int rbase = m0 + waveM * 32 + mi * 16 + khalf * 8;
            #pragma unroll
            for (int v = 0; v < 8; ++v) {
                float val = acc[mi][ni][v] + bv;
                size_t idx = (size_t)(rbase + v) * N + col;
                if constexpr (STORE_F16) ((_Float16*)Cout)[idx] = (_Float16)val;
                else                     ((float*)Cout)[idx]    = val;
            }
        }
    }
}

// ---------------------------------------------------------------------------
// Deformable sampling + softmax. One block per (b, t, head); 64 threads = hd.
// H=14, W=1 level-0 grid: ix = ref_x + off_x - 0.5 ; iy = off_y - 0.5.
// ---------------------------------------------------------------------------
__global__ void msdeform_sample_kernel(const _Float16* __restrict__ value,
                                       const float* __restrict__ offs,
                                       const float* __restrict__ logits,
                                       _Float16* __restrict__ out)
{
    const int gid = blockIdx.x;          // b*T*H + t*H + h
    const int h  = gid % NHEADS;
    const int bt = gid / NHEADS;
    const int t  = bt % T_LEN;
    const int b  = bt / T_LEN;
    const int d  = threadIdx.x;          // 0..63

    const float* op = offs   + ((size_t)bt * NHEADS + h) * (NPOINTS * 2);
    const float* lg = logits + ((size_t)bt * NHEADS + h) * NPOINTS;

    float l0 = lg[0], l1 = lg[1], l2 = lg[2], l3 = lg[3];
    float mx = fmaxf(fmaxf(l0, l1), fmaxf(l2, l3));
    float e0 = expf(l0 - mx), e1 = expf(l1 - mx),
          e2 = expf(l2 - mx), e3 = expf(l3 - mx);
    float inv = 1.0f / (e0 + e1 + e2 + e3);
    float aw[4] = { e0 * inv, e1 * inv, e2 * inv, e3 * inv };

    const float refx = (float)t * (1.0f / (float)(T_LEN - 1));
    float acc = 0.0f;
    #pragma unroll
    for (int p = 0; p < NPOINTS; ++p) {
        float ox = op[2 * p + 0];
        float oy = op[2 * p + 1];
        float ix = refx + ox - 0.5f;     // loc_x * W - 0.5, W = 1
        float iy = oy - 0.5f;            // (oy/H) * H - 0.5
        float x0 = floorf(ix), y0 = floorf(iy);
        float fx = ix - x0,    fy = iy - y0;
        int x0i = (int)x0, y0i = (int)y0;
        float wp = aw[p];
        #pragma unroll
        for (int corner = 0; corner < 4; ++corner) {
            int dx = corner & 1, dy = corner >> 1;
            int xx = x0i + dx, yy = y0i + dy;
            float cw = (dx ? fx : 1.0f - fx) * (dy ? fy : 1.0f - fy);
            if (xx == 0 && yy >= 0 && yy < T_LEN) {   // W=1 => only column 0 valid
                acc += wp * cw *
                    (float)value[(((size_t)b * T_LEN + yy) * NHEADS + h) * HD + d];
            }
        }
    }
    out[((size_t)bt * NHEADS + h) * HD + d] = (_Float16)acc;
}

// ---------------------------------------------------------------------------
extern "C" void kernel_launch(void* const* d_in, const int* in_sizes, int n_in,
                              void* d_out, int out_size, void* d_ws, size_t ws_size,
                              hipStream_t stream)
{
    (void)in_sizes; (void)n_in; (void)out_size; (void)ws_size;

    const float* x       = (const float*)d_in[0];
    const float* W_value = (const float*)d_in[1];
    const float* b_value = (const float*)d_in[2];
    const float* W_off   = (const float*)d_in[3];
    const float* b_off   = (const float*)d_in[4];
    const float* W_attn  = (const float*)d_in[5];
    const float* b_attn  = (const float*)d_in[6];
    const float* W_out   = (const float*)d_in[7];
    const float* b_out   = (const float*)d_in[8];
    const float* W_proj  = (const float*)d_in[9];
    const float* b_proj  = (const float*)d_in[10];

    char* ws = (char*)d_ws;
    size_t cur = 0;
    auto alloc = [&](size_t bytes) -> void* {
        cur = (cur + 255) & ~(size_t)255;
        void* p = ws + cur;
        cur += bytes;
        return p;
    };

    // xh doubles as sampled-output buffer; valh doubles as y buffer (stream order)
    _Float16* xh     = (_Float16*)alloc((size_t)M_ROWS * DM * 2);
    _Float16* valh   = (_Float16*)alloc((size_t)M_ROWS * DM * 2);
    _Float16* WvT    = (_Float16*)alloc((size_t)DM * DM * 2);
    _Float16* WoffT  = (_Float16*)alloc((size_t)DM * (NHEADS * NPOINTS * 2) * 2);
    _Float16* WatT   = (_Float16*)alloc((size_t)DM * (NHEADS * NPOINTS) * 2);
    _Float16* WoT    = (_Float16*)alloc((size_t)DM * DM * 2);
    _Float16* WpT    = (_Float16*)alloc((size_t)DM * DM * 2);
    float* offsb     = (float*)alloc((size_t)M_ROWS * (NHEADS * NPOINTS * 2) * 4);
    float* logitb    = (float*)alloc((size_t)M_ROWS * (NHEADS * NPOINTS) * 4);

    const int NOFF = NHEADS * NPOINTS * 2;   // 128
    const int NATT = NHEADS * NPOINTS;       // 64

    // x -> f16 (row-major, A operand)
    {
        int n = M_ROWS * DM;
        f32_to_f16_kernel<<<(n + 255) / 256, 256, 0, stream>>>(x, xh, n);
    }
    // weights [K,N] -> f16 transposed [N,K] (B operand, K-major)
    auto tcvt = [&](const float* s, _Float16* dptr, int K, int N) {
        transpose_cvt_kernel<<<dim3(N / 32, K / 32), dim3(32, 8), 0, stream>>>(s, dptr, K, N);
    };
    tcvt(W_value, WvT,   DM, DM);
    tcvt(W_off,   WoffT, DM, NOFF);
    tcvt(W_attn,  WatT,  DM, NATT);
    tcvt(W_out,   WoT,   DM, DM);
    tcvt(W_proj,  WpT,   DM, DM);

    dim3 blk(256);

    // value = x @ W_value + b_value   (f16 out)
    wmma_gemm_kernel<true><<<dim3(DM / BN, M_ROWS / BM), blk, 0, stream>>>(
        xh, WvT, b_value, valh, M_ROWS, DM, DM);
    // off = x @ W_off + b_off         (f32 out)
    wmma_gemm_kernel<false><<<dim3(NOFF / BN, M_ROWS / BM), blk, 0, stream>>>(
        xh, WoffT, b_off, offsb, M_ROWS, NOFF, DM);
    // logits = x @ W_attn + b_attn    (f32 out)
    wmma_gemm_kernel<false><<<dim3(NATT / BN, M_ROWS / BM), blk, 0, stream>>>(
        xh, WatT, b_attn, logitb, M_ROWS, NATT, DM);

    // softmax + bilinear gather -> sampled (reuses xh)
    msdeform_sample_kernel<<<B_CLIPS * T_LEN * NHEADS, 64, 0, stream>>>(
        valh, offsb, logitb, xh);

    // y = sampled @ W_out + b_out     (f16 out, reuses valh)
    wmma_gemm_kernel<true><<<dim3(DM / BN, M_ROWS / BM), blk, 0, stream>>>(
        xh, WoT, b_out, valh, M_ROWS, DM, DM);
    // out = y @ W_proj + b_proj       (f32 out -> d_out)
    wmma_gemm_kernel<false><<<dim3(DM / BN, M_ROWS / BM), blk, 0, stream>>>(
        valh, WpT, b_proj, d_out, M_ROWS, DM, DM);
}
